// Pointnet_SA_24378234372448
// MI455X (gfx1250) — matrix operand order repaired and checked by reading, not compiled, not run on hardware
//
#include <hip/hip_runtime.h>
#include <hip/hip_bf16.h>

typedef __attribute__((ext_vector_type(16))) _Float16 v16h;
typedef __attribute__((ext_vector_type(8)))  _Float16 v8h;
typedef __attribute__((ext_vector_type(8)))  float    v8f;

#define NPOINT   1024
#define NSAMPLE  32
#define RADIUS2  0.04f
#define BB       8
#define NN       8192
#define CC       64
#define CPW      4     // centroids per wave in MLP kernel

// ---------------------------------------------------------------------------
// Kernel 1: farthest point sampling. One 1024-thread block per batch; each
// thread owns 8 points in registers. Writes new_xyz directly (fps_idx is only
// an intermediate in the reference).
// ---------------------------------------------------------------------------
__global__ __launch_bounds__(1024) void fps_kernel(const float* __restrict__ xyz,
                                                   float* __restrict__ new_xyz)
{
    const int b    = blockIdx.x;
    const int tid  = threadIdx.x;
    const int lane = tid & 31;
    const int wid  = tid >> 5;
    const float* xb = xyz + (size_t)b * NN * 3;

    float px[8], py[8], pz[8], md[8];
#pragma unroll
    for (int i = 0; i < 8; ++i) {
        int p = tid * 8 + i;
        px[i] = xb[p * 3 + 0];
        py[i] = xb[p * 3 + 1];
        pz[i] = xb[p * 3 + 2];
        md[i] = 1e10f;
    }

    __shared__ float s_c[3];
    __shared__ float s_val[32];
    __shared__ int   s_idx[32];
    __shared__ int   s_far;

    int farthest = 0;
    for (int it = 0; it < NPOINT; ++it) {
        if (tid == (farthest >> 3)) {
            int i = farthest & 7;
            s_c[0] = px[i]; s_c[1] = py[i]; s_c[2] = pz[i];
            float* o = new_xyz + ((size_t)b * NPOINT + it) * 3;
            o[0] = px[i]; o[1] = py[i]; o[2] = pz[i];
        }
        __syncthreads();
        float cx = s_c[0], cy = s_c[1], cz = s_c[2];

        float best = -1.0f; int bidx = 0;
#pragma unroll
        for (int i = 0; i < 8; ++i) {
            float dx = px[i] - cx, dy = py[i] - cy, dz = pz[i] - cz;
            float d2 = dx * dx + dy * dy + dz * dz;
            md[i] = fminf(md[i], d2);
            if (md[i] > best) { best = md[i]; bidx = tid * 8 + i; }
        }
        // wave32 reduction (max value, tie -> lower index, matching argmax)
#pragma unroll
        for (int off = 16; off > 0; off >>= 1) {
            float ov = __shfl_down(best, off, 32);
            int   oi = __shfl_down(bidx, off, 32);
            if (ov > best || (ov == best && oi < bidx)) { best = ov; bidx = oi; }
        }
        if (lane == 0) { s_val[wid] = best; s_idx[wid] = bidx; }
        __syncthreads();
        if (wid == 0) {
            best = s_val[lane]; bidx = s_idx[lane];
#pragma unroll
            for (int off = 16; off > 0; off >>= 1) {
                float ov = __shfl_down(best, off, 32);
                int   oi = __shfl_down(bidx, off, 32);
                if (ov > best || (ov == best && oi < bidx)) { best = ov; bidx = oi; }
            }
            if (lane == 0) s_far = bidx;
        }
        __syncthreads();
        farthest = s_far;
    }
}

// ---------------------------------------------------------------------------
// Kernel 2: ball query. One wave per centroid; ballot + prefix popcount gives
// first-NSAMPLE-in-index-order (== sort semantics of the reference).
// ---------------------------------------------------------------------------
__global__ __launch_bounds__(256) void ballq_kernel(const float* __restrict__ xyz,
                                                    const float* __restrict__ new_xyz,
                                                    int* __restrict__ idx_out)
{
    const int gw   = (int)((blockIdx.x * blockDim.x + threadIdx.x) >> 5);
    const int lane = threadIdx.x & 31;
    const int b = gw / NPOINT;
    const int s = gw % NPOINT;

    const float* xb = xyz + (size_t)b * NN * 3;
    const float* c  = new_xyz + ((size_t)b * NPOINT + s) * 3;
    const float cx = c[0], cy = c[1], cz = c[2];
    int* out = idx_out + ((size_t)b * NPOINT + s) * NSAMPLE;

    int cnt = 0, first = -1;
    for (int base = 0; base < NN && cnt < NSAMPLE; base += 32) {
        int p = base + lane;
        float dx = xb[p * 3 + 0] - cx;
        float dy = xb[p * 3 + 1] - cy;
        float dz = xb[p * 3 + 2] - cz;
        bool in = (dx * dx + dy * dy + dz * dz) <= RADIUS2;
        unsigned mask = (unsigned)__ballot(in);
        if (mask) {
            if (first < 0) first = base + __builtin_ctz(mask);
            int pos = cnt + __popc(mask & ((1u << lane) - 1u));
            if (in && pos < NSAMPLE) out[pos] = p;
            cnt += __popc(mask);
        }
    }
    if (cnt < NSAMPLE) {
        if (first < 0) first = 0;
        if (lane >= cnt) out[lane] = first;
    }
}

// ---------------------------------------------------------------------------
// Kernel 3: weight prep. Transpose + pad + convert to f16 once:
//   W0t[64][96] (K padded 67->96) | W1t[64][64] | W2t[128][64], stored [n][k]
// so B fragments become contiguous 16B LDS reads.
// ---------------------------------------------------------------------------
__global__ void wprep_kernel(const float* __restrict__ W0,
                             const float* __restrict__ W1,
                             const float* __restrict__ W2,
                             _Float16* __restrict__ wt)
{
    const int total = 64 * 96 + 64 * 64 + 128 * 64;
    int t = blockIdx.x * blockDim.x + threadIdx.x;
    if (t >= total) return;
    if (t < 64 * 96) {
        int n = t / 96, k = t % 96;
        wt[t] = (k < 67) ? (_Float16)W0[k * 64 + n] : (_Float16)0.0f;
    } else if (t < 64 * 96 + 64 * 64) {
        int u = t - 64 * 96; int n = u / 64, k = u % 64;
        wt[t] = (_Float16)W1[k * 64 + n];
    } else {
        int u = t - 64 * 96 - 64 * 64; int n = u / 64, k = u % 64;
        wt[t] = (_Float16)W2[k * 128 + n];
    }
}

// ---------------------------------------------------------------------------
// Kernel 4: fused gather + 3-layer MLP (WMMA f16 -> f32 acc) + maxpool.
// 2 waves/block, one centroid per wave at a time, CPW centroids per wave.
// A fragments hoisted to registers per layer; biases preloaded per lane.
// ---------------------------------------------------------------------------
__global__ __launch_bounds__(64) void mlp_kernel(const float* __restrict__ xyz,
                                                 const float* __restrict__ points,
                                                 const float* __restrict__ new_xyz,
                                                 const int* __restrict__ idx,
                                                 const _Float16* __restrict__ wt,
                                                 const float* __restrict__ b0,
                                                 const float* __restrict__ b1,
                                                 const float* __restrict__ b2,
                                                 float* __restrict__ out_points)
{
    __shared__ alignas(32) _Float16 sW[64 * 96 + 64 * 64 + 128 * 64]; // 36 KB
    __shared__ alignas(32) _Float16 hA[2][32 * 96];                   // 12 KB
    __shared__ alignas(32) _Float16 hB[2][32 * 64];                   //  8 KB

    const int tid  = threadIdx.x;
    const int lane = tid & 31;
    const int wid  = tid >> 5;

    // cooperative weight stage (f16 pairs as u32)
    {
        const uint32_t* src = (const uint32_t*)wt;
        uint32_t* dst = (uint32_t*)sW;
        const int nw = (64 * 96 + 64 * 64 + 128 * 64) / 2;
        for (int i = tid; i < nw; i += 64) dst[i] = src[i];
    }
    __syncthreads();

    const _Float16* sW0 = sW;
    const _Float16* sW1 = sW + 64 * 96;
    const _Float16* sW2 = sW + 64 * 96 + 64 * 64;

    // per-lane bias registers (N = lane&15 within each n-tile)
    float bias0[4], bias1[4], bias2[8];
#pragma unroll
    for (int nt = 0; nt < 4; ++nt) bias0[nt] = b0[nt * 16 + (lane & 15)];
#pragma unroll
    for (int nt = 0; nt < 4; ++nt) bias1[nt] = b1[nt * 16 + (lane & 15)];
#pragma unroll
    for (int nt = 0; nt < 8; ++nt) bias2[nt] = b2[nt * 16 + (lane & 15)];

    // A fragment: 16x32 f16, lane<16 holds M=lane, K = kb..kb+7 & kb+16..kb+23
    // with kb = kt*32 (+8 for lanes 16-31)
    auto loadA = [&](const _Float16* buf, int stride, int mt, int kt) -> v16h {
        int m  = (lane & 15) + 16 * mt;
        int kb = kt * 32 + ((lane >> 4) << 3);
        const _Float16* rp = buf + m * stride + kb;
        v8h lo = *(const v8h*)(rp);
        v8h hi = *(const v8h*)(rp + 16);
        v16h a;
#pragma unroll
        for (int j = 0; j < 8; ++j) { a[j] = lo[j]; a[j + 8] = hi[j]; }
        return a;
    };
    // B fragment: 32x16 f16, lane<16 holds N=lane, K = kb..kb+7 & kb+16..kb+23
    // with kb = kt*32 (+8 for lanes 16-31)
    auto loadB = [&](const _Float16* w, int kstride, int kt, int nt) -> v16h {
        int n  = (lane & 15) + 16 * nt;
        int kb = kt * 32 + ((lane >> 4) << 3);
        const _Float16* rp = w + n * kstride + kb;
        v8h lo = *(const v8h*)(rp);
        v8h hi = *(const v8h*)(rp + 16);
        v16h bv;
#pragma unroll
        for (int j = 0; j < 8; ++j) { bv[j] = lo[j]; bv[j + 8] = hi[j]; }
        return bv;
    };
    // C/D: VGPR r holds M = r (+8 for lanes 16-31), N = lane&15
    auto storeD = [&](_Float16* buf, int stride, int nt, v8f a0, v8f a1, float bias) {
        int n  = nt * 16 + (lane & 15);
        int mh = ((lane >> 4) & 1) * 8;
#pragma unroll
        for (int r = 0; r < 8; ++r) {
            float v0 = fmaxf(a0[r] + bias, 0.0f);
            float v1 = fmaxf(a1[r] + bias, 0.0f);
            buf[(mh + r) * stride + n]      = (_Float16)v0;
            buf[(16 + mh + r) * stride + n] = (_Float16)v1;
        }
    };

    const int gw = blockIdx.x * 2 + wid;
    for (int c = 0; c < CPW; ++c) {
        const int sg = gw * CPW + c;           // 0 .. B*NPOINT-1
        const int b  = sg / NPOINT;
        const int* ip = idx + (size_t)sg * NSAMPLE;
        const float* cen = new_xyz + (size_t)sg * 3;
        const float cx = cen[0], cy = cen[1], cz = cen[2];

        _Float16* h0 = hA[wid];
        // ---- gather h0 = [rel_xyz(3) | point_feats(64) | pad->96] as f16 ----
        for (int m = 0; m < NSAMPLE; ++m) {
            int pidx = ip[m];
            const float* pr = points + ((size_t)b * NN + pidx) * CC;
            const float* xr = xyz + ((size_t)b * NN + pidx) * 3;
            float v;
            if (lane < 3) v = xr[lane] - ((lane == 0) ? cx : (lane == 1) ? cy : cz);
            else          v = pr[lane - 3];
            h0[m * 96 + lane]      = (_Float16)v;
            h0[m * 96 + 32 + lane] = (_Float16)pr[lane + 29];
            int k3 = 64 + lane;
            h0[m * 96 + k3] = (k3 < 67) ? (_Float16)pr[k3 - 3] : (_Float16)0.0f;
        }

        // ---- layer 0: [32x96] @ [96x64] -> hB ----
        {
            v16h A[3][2];
#pragma unroll
            for (int kt = 0; kt < 3; ++kt) {
                A[kt][0] = loadA(h0, 96, 0, kt);
                A[kt][1] = loadA(h0, 96, 1, kt);
            }
#pragma unroll
            for (int nt = 0; nt < 4; ++nt) {
                v8f a0 = {}, a1 = {};
#pragma unroll
                for (int kt = 0; kt < 3; ++kt) {
                    v16h bf = loadB(sW0, 96, kt, nt);
                    a0 = __builtin_amdgcn_wmma_f32_16x16x32_f16(false, A[kt][0], false, bf, (short)0, a0, false, false);
                    a1 = __builtin_amdgcn_wmma_f32_16x16x32_f16(false, A[kt][1], false, bf, (short)0, a1, false, false);
                }
                storeD(hB[wid], 64, nt, a0, a1, bias0[nt]);
            }
        }

        // ---- layer 1: [32x64] @ [64x64] -> hA (stride 96, k<64 used) ----
        {
            v16h A[2][2];
#pragma unroll
            for (int kt = 0; kt < 2; ++kt) {
                A[kt][0] = loadA(hB[wid], 64, 0, kt);
                A[kt][1] = loadA(hB[wid], 64, 1, kt);
            }
#pragma unroll
            for (int nt = 0; nt < 4; ++nt) {
                v8f a0 = {}, a1 = {};
#pragma unroll
                for (int kt = 0; kt < 2; ++kt) {
                    v16h bf = loadB(sW1, 64, kt, nt);
                    a0 = __builtin_amdgcn_wmma_f32_16x16x32_f16(false, A[kt][0], false, bf, (short)0, a0, false, false);
                    a1 = __builtin_amdgcn_wmma_f32_16x16x32_f16(false, A[kt][1], false, bf, (short)0, a1, false, false);
                }
                storeD(hA[wid], 96, nt, a0, a1, bias1[nt]);
            }
        }

        // ---- layer 2: [32x64] @ [64x128] + ReLU + max over 32 samples ----
        {
            v16h A[2][2];
#pragma unroll
            for (int kt = 0; kt < 2; ++kt) {
                A[kt][0] = loadA(hA[wid], 96, 0, kt);
                A[kt][1] = loadA(hA[wid], 96, 1, kt);
            }
            float* op = out_points + (size_t)sg * 128;
#pragma unroll
            for (int nt = 0; nt < 8; ++nt) {
                v8f a0 = {}, a1 = {};
#pragma unroll
                for (int kt = 0; kt < 2; ++kt) {
                    v16h bf = loadB(sW2, 64, kt, nt);
                    a0 = __builtin_amdgcn_wmma_f32_16x16x32_f16(false, A[kt][0], false, bf, (short)0, a0, false, false);
                    a1 = __builtin_amdgcn_wmma_f32_16x16x32_f16(false, A[kt][1], false, bf, (short)0, a1, false, false);
                }
                float bias = bias2[nt];
                float mx = 0.0f;  // ReLU floor: max(relu(x)) == max(max(x),0)
#pragma unroll
                for (int r = 0; r < 8; ++r) {
                    mx = fmaxf(mx, a0[r] + bias);
                    mx = fmaxf(mx, a1[r] + bias);
                }
                mx = fmaxf(mx, __shfl_xor(mx, 16, 32));  // combine M-halves (same N)
                if (lane < 16) op[nt * 16 + lane] = mx;
            }
        }
    }
}

// ---------------------------------------------------------------------------
extern "C" void kernel_launch(void* const* d_in, const int* in_sizes, int n_in,
                              void* d_out, int out_size, void* d_ws, size_t ws_size,
                              hipStream_t stream)
{
    const float* xyz    = (const float*)d_in[0];
    const float* points = (const float*)d_in[1];
    const float* W0 = (const float*)d_in[2];
    const float* b0 = (const float*)d_in[3];
    const float* W1 = (const float*)d_in[4];
    const float* b1 = (const float*)d_in[5];
    const float* W2 = (const float*)d_in[6];
    const float* b2 = (const float*)d_in[7];

    float* out        = (float*)d_out;
    float* new_xyz    = out;                           // 8*1024*3
    float* new_points = out + (size_t)BB * NPOINT * 3; // 8*1024*128

    char* ws = (char*)d_ws;
    int*      ballidx = (int*)ws;                                   // 1 MB
    _Float16* wt      = (_Float16*)(ws + (size_t)BB * NPOINT * NSAMPLE * 4);

    fps_kernel<<<BB, 1024, 0, stream>>>(xyz, new_xyz);
    wprep_kernel<<<(64 * 96 + 64 * 64 + 128 * 64 + 255) / 256, 256, 0, stream>>>(W0, W1, W2, wt);
    ballq_kernel<<<(BB * NPOINT) / 8, 256, 0, stream>>>(xyz, new_xyz, ballidx);
    mlp_kernel<<<(BB * NPOINT) / (2 * CPW), 64, 0, stream>>>(xyz, points, new_xyz, ballidx,
                                                             wt, b0, b1, b2, new_points);
}
